// Net_86535001080079
// MI455X (gfx1250) — matrix operand order repaired and checked by reading, not compiled, not run on hardware
//
#include <hip/hip_runtime.h>
#include <stdint.h>

#define N_NODES  131072
#define N_EDGES  262144
#define N_GRAPHS 4096

typedef __attribute__((ext_vector_type(16))) __bf16 v16bf;
typedef __attribute__((ext_vector_type(8)))  float  v8f;

struct Seg { const unsigned short* p; const int* idx; int ld; int w; };

union Frag { v16bf v; uint4 u[2]; unsigned short s[16]; };

__device__ __forceinline__ unsigned short f2bf(float f) {
  unsigned u = __float_as_uint(f);
  u += 0x7fffu + ((u >> 16) & 1u);            // round-to-nearest-even
  return (unsigned short)(u >> 16);
}
__device__ __forceinline__ float elu(float v) { return v > 0.f ? v : __expf(v) - 1.f; }

// One segment-contiguous slice of GEMM-1: A chunks stream from a fixed per-lane
// row pointer (segment widths padded to 32 -> no chunk ever crosses a segment).
template<int NH>
__device__ __forceinline__ void gemm1_seg(v8f (&acc)[NH], const unsigned short* base,
                                          int ktiles, int& kt,
                                          const unsigned short* W1l, int hi8) {
  for (int t = 0; t < ktiles; ++t, ++kt) {
    Frag a;
    const unsigned short* pa = base + t * 32 + hi8;
    a.u[0] = *(const uint4*)pa;
    a.u[1] = *(const uint4*)(pa + 16);
    if (NH <= 4) {                               // constant-folds per instantiation
      Frag bf[NH];
#pragma unroll
      for (int h = 0; h < NH; ++h) {
        const unsigned short* q = W1l + (size_t)(kt * NH + h) * 512;
        bf[h].u[0] = *(const uint4*)q;
        bf[h].u[1] = *(const uint4*)(q + 8);
      }
#pragma unroll
      for (int h = 0; h < NH; ++h)
        acc[h] = __builtin_amdgcn_wmma_f32_16x16x32_bf16(false, a.v, false, bf[h].v,
                                                         (short)0, acc[h], false, false);
    } else {                                     // NH=16: stream B to bound registers
#pragma unroll
      for (int h = 0; h < NH; ++h) {
        const unsigned short* q = W1l + (size_t)(kt * NH + h) * 512;
        Frag b;
        b.u[0] = *(const uint4*)q;
        b.u[1] = *(const uint4*)(q + 8);
        acc[h] = __builtin_amdgcn_wmma_f32_16x16x32_bf16(false, a.v, false, b.v,
                                                         (short)0, acc[h], false, false);
      }
    }
  }
}

// Fused 2-layer MLP: out = (ELU(A @ W1 + b1)) @ W2 + b2
// A = concat of up to 3 gathered bf16 segments, each padded to a multiple of 32
// columns (weight pack inserts matching zero rows). One wave per 16-row strip;
// hidden strip lives in wave-private LDS. Epilogue: bf16 / fp32 store, scatter-add.
template<int NH>
__global__ __launch_bounds__(128) void mlp2_kernel(
    Seg s0, Seg s1, Seg s2, int M,
    const unsigned short* __restrict__ W1p, const float* __restrict__ b1,
    const unsigned short* __restrict__ W2p, const float* __restrict__ b2,
    int N2,
    unsigned short* __restrict__ outb, int ldo,
    float* __restrict__ outf, int ldof,
    float* __restrict__ scat, const int* __restrict__ sidx) {
  __shared__ __align__(16) unsigned short hsh[4][NH * 256];  // 4 waves x (16 rows x NH*16)
  const int wave  = threadIdx.x >> 5;
  const int lane  = threadIdx.x & 31;
  const int strip = blockIdx.x * 4 + wave;
  if (strip * 16 >= M) return;                 // never taken: M % 64 == 0 for all calls
  const int hi8   = (lane >> 4) * 8;           // half-wave K sub-chunk / +8 row offset
  const int colb  = lane & 15;
  const int mrow  = strip * 16 + colb;         // A row owned by this lane

  const v8f vzero = {0.f, 0.f, 0.f, 0.f, 0.f, 0.f, 0.f, 0.f};
  v8f acc[NH];
#pragma unroll
  for (int h = 0; h < NH; ++h) acc[h] = vzero;

  const unsigned short* W1l = W1p + lane * 16;
  // Resolve gather indices ONCE, then run straight-line per-segment k-loops.
  int kt = 0;
  if (s0.w) {
    int r = s0.idx ? s0.idx[mrow] : mrow;
    gemm1_seg<NH>(acc, s0.p + (size_t)r * s0.ld, ((s0.w + 31) & ~31) >> 5, kt, W1l, hi8);
  }
  if (s1.w) {
    int r = s1.idx ? s1.idx[mrow] : mrow;
    gemm1_seg<NH>(acc, s1.p + (size_t)r * s1.ld, ((s1.w + 31) & ~31) >> 5, kt, W1l, hi8);
  }
  if (s2.w) {
    int r = s2.idx ? s2.idx[mrow] : mrow;
    gemm1_seg<NH>(acc, s2.p + (size_t)r * s2.ld, ((s2.w + 31) & ~31) >> 5, kt, W1l, hi8);
  }

  // ---- bias + ELU -> bf16 hidden strip in wave-private LDS ----
  unsigned short* hb = hsh[wave];
#pragma unroll
  for (int h = 0; h < NH; ++h) {
    const float bias = b1[h * 16 + colb];
#pragma unroll
    for (int r = 0; r < 8; ++r) {
      const float v = elu(acc[h][r] + bias);   // C layout: row r(+8 for hi), col = colb
      hb[(r + hi8) * (NH * 16) + h * 16 + colb] = f2bf(v);
    }
  }
  // DS ops within a wave are in-order; region is wave-private -> no barrier needed.

  // ---- GEMM 2: out = hidden @ W2, streamed one 16-col tile at a time ----
  const unsigned short* hr  = hb + colb * (NH * 16);
  const unsigned short* W2l = W2p + lane * 16;
  const int nt2 = N2 >> 4;
  for (int nt = 0; nt < nt2; ++nt) {
    v8f c2 = vzero;
#pragma unroll
    for (int k2 = 0; k2 < NH / 2; ++k2) {
      Frag a2;
      const int c0 = k2 * 32 + hi8;
      a2.u[0] = *(const uint4*)(hr + c0);
      a2.u[1] = *(const uint4*)(hr + c0 + 16);
      Frag b2f;
      const unsigned short* q = W2l + (size_t)(k2 * nt2 + nt) * 512;
      b2f.u[0] = *(const uint4*)q;
      b2f.u[1] = *(const uint4*)(q + 8);
      c2 = __builtin_amdgcn_wmma_f32_16x16x32_bf16(false, a2.v, false, b2f.v,
                                                   (short)0, c2, false, false);
    }
    const float bias = b2[nt * 16 + colb];
#pragma unroll
    for (int r = 0; r < 8; ++r) {
      const float v = c2[r] + bias;
      const int mg = strip * 16 + r + hi8;
      const int c  = nt * 16 + colb;
      if (outb) outb[(size_t)mg * ldo + c] = f2bf(v);
      if (outf) outf[(size_t)mg * ldof + c] = v;
      if (scat) atomicAdd(scat + (size_t)sidx[mg] * N2 + c, v);   // fused segment-sum
    }
  }
}

// Repack fp32 W[K x N] into WMMA B fragments with *segment-aware* K padding:
// virtual K axis = [pad32(w0) | pad32(w1) | pad32(w2)], zero rows in the pads.
// frag f = kt*(N/16)+nt: lane holds col nt*16+(lane&15), K = kt*32+(j<8 ? 8h+j : 16+8h+j-8), h=lane>=16.
__global__ void pack_w_kernel(const float* __restrict__ W, int N,
                              int w0, int w1, int w2,
                              unsigned short* __restrict__ out) {
  const int wp0 = (w0 + 31) & ~31, wp1 = (w1 + 31) & ~31, wp2 = (w2 + 31) & ~31;
  const long total = (long)((wp0 + wp1 + wp2) / 32) * (N / 16) * 512;
  for (long i = (long)blockIdx.x * blockDim.x + threadIdx.x; i < total;
       i += (long)gridDim.x * blockDim.x) {
    const long f    = i >> 9;
    const int  rem  = (int)(i & 511);
    const int  lane = rem >> 4, j = rem & 15;
    const int  ntiles = N / 16;
    const int  kt = (int)(f / ntiles), nt = (int)(f % ntiles);
    const int  c = kt * 32 + ((lane >= 16) ? 8 : 0) + (j < 8 ? j : j + 8);   // virtual (padded) K
    const int  n = nt * 16 + (lane & 15);
    int r; bool valid;
    if (c < wp0)            { r = c;                 valid = c < w0; }
    else if (c < wp0 + wp1) { int cc = c - wp0;      r = w0 + cc;      valid = cc < w1; }
    else                    { int cc = c - wp0 - wp1; r = w0 + w1 + cc; valid = cc < w2; }
    out[i] = f2bf(valid ? W[(size_t)r * N + n] : 0.f);
  }
}

// Column sums and sum-of-squares (for training-mode BN). out[0..C)=sum, out[C..2C)=sumsq.
__global__ void colstats_kernel(const float* __restrict__ v, int R, int C,
                                float* __restrict__ out) {
  const int c = threadIdx.x % C;
  const int r0 = (int)((blockIdx.x * blockDim.x + threadIdx.x) / C);
  const int rs = (int)((gridDim.x * blockDim.x) / C);
  float s = 0.f, q = 0.f;
  for (int r = r0; r < R; r += rs) { const float x = v[(size_t)r * C + c]; s += x; q += x * x; }
  atomicAdd(&out[c], s);
  atomicAdd(&out[C + c], q);
}

__global__ void bn_apply_kernel(const float* __restrict__ v, const float* __restrict__ st,
                                const float* __restrict__ g, const float* __restrict__ b,
                                int R, int C, unsigned short* __restrict__ out, int ldo) {
  const long i = (long)blockIdx.x * blockDim.x + threadIdx.x;
  if (i >= (long)R * C) return;
  const int c = (int)(i % C); const long r = i / C;
  const float m = st[c] / (float)R;
  const float var = st[C + c] / (float)R - m * m;
  const float y = (v[i] - m) * rsqrtf(var + 1e-5f) * g[c] + b[c];
  out[r * ldo + c] = f2bf(y);
}

__global__ void count_kernel(const int* __restrict__ idx, int n, float* __restrict__ cnt) {
  const int i = blockIdx.x * blockDim.x + threadIdx.x;
  if (i < n) atomicAdd(&cnt[idx[i]], 1.f);
}

__global__ void mean_bf_kernel(const float* __restrict__ sums, const float* __restrict__ cnt,
                               int R, int C, unsigned short* __restrict__ out) {
  const long i = (long)blockIdx.x * blockDim.x + threadIdx.x;
  if (i >= (long)R * C) return;
  const long r = i / C;
  float d = cnt[r]; d = d > 1.f ? d : 1.f;
  out[i] = f2bf(sums[i] / d);
}

extern "C" void kernel_launch(void* const* d_in, const int* in_sizes, int n_in,
                              void* d_out, int out_size, void* d_ws, size_t ws_size,
                              hipStream_t stream) {
  (void)in_sizes; (void)n_in; (void)out_size; (void)ws_size;
  // --- input leaf order (setup_inputs insertion order, depth-first) ---
  const float* x_in  = (const float*)d_in[0];
  const float* e_in  = (const float*)d_in[1];
  const int*   ei    = (const int*)d_in[2];
  const int*   row   = ei;
  const int*   col   = ei + N_EDGES;
  const int*   batch = (const int*)d_in[3];
  auto F = [&](int i) { return (const float*)d_in[i]; };
  // 4..7: bn_node.{g,b}, bn_edge.{g,b}; meta i leaves at 8+i*16; lin1/lin2 at 120..123

  static const int NF[7] = {64, 32, 32, 32, 32, 32, 32};
  static const int EF[7] = {16, 256, 256, 256, 256, 256, 256};
  static const int EO[7] = {256, 256, 256, 256, 256, 256, 32};
  static const int NO[7] = {32, 32, 32, 32, 32, 32, 32};
  static const int GP[7] = {0, 32, 32, 32, 32, 32, 32};
  static const int GO[7] = {32, 32, 32, 32, 32, 32, 64};

  // --- workspace carve-out ---
  char* wsb = (char*)d_ws; size_t off = 0;
  auto alloc = [&](size_t bytes) -> void* {
    void* p = wsb + off; off = (off + bytes + 255) & ~(size_t)255; return p;
  };
  auto packW3 = [&](const float* W, int w0, int w1, int w2, int N) -> const unsigned short* {
    const int Kp = ((w0 + 31) & ~31) + ((w1 + 31) & ~31) + ((w2 + 31) & ~31);
    const size_t halves = (size_t)(Kp / 32) * (N / 16) * 512;
    unsigned short* p = (unsigned short*)alloc(halves * 2);
    int blocks = (int)((halves + 255) / 256); if (blocks > 2048) blocks = 2048;
    pack_w_kernel<<<blocks, 256, 0, stream>>>(W, N, w0, w1, w2, p);
    return p;
  };

  // weight packs (bf16 B-fragment layout; <2MB total, L2-resident)
  const unsigned short *pe0[7], *pe1[7], *pn10[7], *pn11[7], *pn20[7], *pn21[7], *pg0[7], *pg1[7];
  for (int i = 0; i < 7; ++i) {
    const int bi = 8 + i * 16;
    pe0[i]  = packW3(F(bi + 0),  NF[i], NF[i], EF[i], 32);
    pe1[i]  = packW3(F(bi + 2),  32, 0, 0, EO[i]);
    pn10[i] = packW3(F(bi + 4),  NF[i], EO[i], 0, 64);
    pn11[i] = packW3(F(bi + 6),  64, 0, 0, 64);
    pn20[i] = packW3(F(bi + 8),  NF[i], 64, 0, 64);
    pn21[i] = packW3(F(bi + 10), 64, 0, 0, NO[i]);
    pg0[i]  = packW3(F(bi + 12), GP[i], NO[i], 0, 64);
    pg1[i]  = packW3(F(bi + 14), 64, 0, 0, GO[i]);
  }
  const unsigned short* pl1 = packW3(F(120), 64, 0, 0, 256);
  const unsigned short* pl2 = packW3(F(122), 256, 0, 0, 2048);

  // activation buffers (bf16 raw ushort)
  unsigned short* xa     = (unsigned short*)alloc((size_t)N_NODES * 64 * 2);
  unsigned short* xb     = (unsigned short*)alloc((size_t)N_NODES * 64 * 2);
  unsigned short* ea     = (unsigned short*)alloc((size_t)N_EDGES * 256 * 2);
  unsigned short* eb     = (unsigned short*)alloc((size_t)N_EDGES * 256 * 2);
  unsigned short* aggbf  = (unsigned short*)alloc((size_t)N_NODES * 64 * 2);
  unsigned short* nmbf   = (unsigned short*)alloc((size_t)N_GRAPHS * 32 * 2);
  unsigned short* ua     = (unsigned short*)alloc((size_t)N_GRAPHS * 64 * 2);
  unsigned short* ub     = (unsigned short*)alloc((size_t)N_GRAPHS * 64 * 2);
  float* aggsum = (float*)alloc((size_t)N_NODES * 64 * 4);
  float* nmsum  = (float*)alloc((size_t)N_GRAPHS * 32 * 4);
  float* cntN   = (float*)alloc((size_t)N_NODES * 4);
  float* cntG   = (float*)alloc((size_t)N_GRAPHS * 4);
  float* stx    = (float*)alloc(128 * 4);
  float* ste    = (float*)alloc(32 * 4);

  const Seg SNULL = {nullptr, nullptr, 0, 0};
  auto launch_mlp = [&](int NH, Seg s0, Seg s1, Seg s2, int M,
                        const unsigned short* W1p, const float* b1,
                        const unsigned short* W2p, const float* b2, int N2,
                        unsigned short* outb, int ldo, float* outf, int ldof,
                        float* scat, const int* sidx) {
    dim3 g(M / 64), b(128);
    switch (NH) {
      case 2:  mlp2_kernel<2><<<g, b, 0, stream>>>(s0, s1, s2, M, W1p, b1, W2p, b2, N2, outb, ldo, outf, ldof, scat, sidx); break;
      case 4:  mlp2_kernel<4><<<g, b, 0, stream>>>(s0, s1, s2, M, W1p, b1, W2p, b2, N2, outb, ldo, outf, ldof, scat, sidx); break;
      case 16: mlp2_kernel<16><<<g, b, 0, stream>>>(s0, s1, s2, M, W1p, b1, W2p, b2, N2, outb, ldo, outf, ldof, scat, sidx); break;
    }
  };

  // --- BatchNorm (training stats) -> bf16 ---
  hipMemsetAsync(stx, 0, 128 * 4, stream);
  hipMemsetAsync(ste, 0, 32 * 4, stream);
  // layer-0 edge features are stored ld=32 with a zeroed pad (16 real cols):
  hipMemsetAsync(ea, 0, (size_t)N_EDGES * 32 * 2, stream);
  colstats_kernel<<<512, 256, 0, stream>>>(x_in, N_NODES, 64, stx);
  colstats_kernel<<<512, 256, 0, stream>>>(e_in, N_EDGES, 16, ste);
  bn_apply_kernel<<<(N_NODES * 64) / 256, 256, 0, stream>>>(x_in, stx, F(4), F(5), N_NODES, 64, xa, 64);
  bn_apply_kernel<<<(N_EDGES * 16) / 256, 256, 0, stream>>>(e_in, ste, F(6), F(7), N_EDGES, 16, ea, 32);

  // segment counts are loop-invariant
  hipMemsetAsync(cntN, 0, (size_t)N_NODES * 4, stream);
  hipMemsetAsync(cntG, 0, (size_t)N_GRAPHS * 4, stream);
  count_kernel<<<N_EDGES / 256, 256, 0, stream>>>(col, N_EDGES, cntN);
  count_kernel<<<N_NODES / 256, 256, 0, stream>>>(batch, N_NODES, cntG);

  unsigned short *xc = xa, *xn = xb, *ec = ea, *en = eb, *uc = ua, *un = ub;
  int ldx = 64, lde = 32;   // layer-0 e rows padded to 32

  for (int i = 0; i < 7; ++i) {
    const int nf = NF[i], ef = EF[i], eo = EO[i], no = NO[i], gp = GP[i], go = GO[i];
    // EdgeModel: ELU-MLP( [x[row], x[col], e] )  -> e' (bf16)
    {
      Seg s0 = {xc, row, ldx, nf}, s1 = {xc, col, ldx, nf}, s2 = {ec, nullptr, lde, ef};
      launch_mlp(2, s0, s1, s2, N_EDGES, pe0[i], F(8 + i * 16 + 1),
                 pe1[i], F(8 + i * 16 + 3), eo, en, eo, nullptr, 0, nullptr, nullptr);
    }
    // NodeModel part 1: h = MLP([x[row], e']); fused scatter-sum(h) over col
    hipMemsetAsync(aggsum, 0, (size_t)N_NODES * 64 * 4, stream);
    {
      Seg s0 = {xc, row, ldx, nf}, s1 = {en, nullptr, eo, eo};
      launch_mlp(4, s0, s1, SNULL, N_EDGES, pn10[i], F(8 + i * 16 + 5),
                 pn11[i], F(8 + i * 16 + 7), 64, nullptr, 0, nullptr, 0, aggsum, col);
    }
    mean_bf_kernel<<<(N_NODES * 64) / 256, 256, 0, stream>>>(aggsum, cntN, N_NODES, 64, aggbf);
    // NodeModel part 2: x' = MLP([x, agg]); fused scatter-sum(x') over batch
    hipMemsetAsync(nmsum, 0, (size_t)N_GRAPHS * 32 * 4, stream);
    {
      Seg s0 = {xc, nullptr, ldx, nf}, s1 = {aggbf, nullptr, 64, 64};
      launch_mlp(4, s0, s1, SNULL, N_NODES, pn20[i], F(8 + i * 16 + 9),
                 pn21[i], F(8 + i * 16 + 11), no, xn, no, nullptr, 0, nmsum, batch);
    }
    mean_bf_kernel<<<(N_GRAPHS * 32) / 256, 256, 0, stream>>>(nmsum, cntG, N_GRAPHS, 32, nmbf);
    // GlobalModel: u' = MLP([u, nm])
    {
      Seg s0 = {uc, nullptr, gp, gp}, s1 = {nmbf, nullptr, 32, 32};
      launch_mlp(4, s0, s1, SNULL, N_GRAPHS, pg0[i], F(8 + i * 16 + 13),
                 pg1[i], F(8 + i * 16 + 15), go, un, go, nullptr, 0, nullptr, nullptr);
    }
    { unsigned short* t = xc; xc = xn; xn = t; }
    { unsigned short* t = ec; ec = en; en = t; }
    { unsigned short* t = uc; uc = un; un = t; }
    ldx = no; lde = eo;
  }

  // head: out = ELU(u @ W1 + b1) @ W2 + b2  (4096 x 2048, fp32 -> d_out)
  {
    Seg s0 = {uc, nullptr, 64, 64};
    launch_mlp(16, s0, SNULL, SNULL, N_GRAPHS, pl1, F(121), pl2, F(123), 2048,
               nullptr, 0, (float*)d_out, 2048, nullptr, nullptr);
  }
}
